// StructLNN_44023414784371
// MI455X (gfx1250) — matrix operand order repaired and checked by reading, not compiled, not run on hardware
//
#include <hip/hip_runtime.h>
#include <hip/hip_bf16.h>

typedef __bf16 bf16;
typedef __attribute__((ext_vector_type(16))) __bf16 v16bf;
typedef __attribute__((ext_vector_type(8)))  __bf16 v8bf;
typedef __attribute__((ext_vector_type(8)))  float  v8f;

#define B_DIM 64
#define T_DIM 4096
#define D_DIM 67
#define H_DIM 128
// concat weight: [256 rows (2H)] x [512 cols (ff1|ff2|ta|tb)]
#define NCOL 512
#define KROW 256

// -------- CDNA5 async global->LDS copy (ASYNCcnt-tracked), ISA 15.18.3 -----
__device__ __forceinline__ void async_load_b64(unsigned lds_addr, const void* gaddr) {
  asm volatile("global_load_async_to_lds_b64 %0, %1, off"
               :: "v"(lds_addr), "v"((unsigned long long)(uintptr_t)gaddr)
               : "memory");
}
__device__ __forceinline__ void wait_async0() {
  asm volatile("s_wait_asynccnt 0x0" ::: "memory");
}

// -------- branch-free transcendentals on v_exp_f32 / v_rcp_f32 (TRANS32) ---
__device__ __forceinline__ float fast_sigmoid(float x) {
  return __builtin_amdgcn_rcpf(1.f + __builtin_amdgcn_exp2f(-1.442695041f * x));
}
__device__ __forceinline__ float fast_tanh(float x) {
  return 2.f * fast_sigmoid(2.f * x) - 1.f;
}

// ---------------------------------------------------------------------------
// Kernel 1: pack the four [256,128] f32 weights into column-major bf16.
// Wt[n][k] = W_concat[k][n] -> each WMMA B-fragment is one contiguous 32B load
// (col = 16*tile + (lane&15), k = 32*kk + 16*(lane>>4)).
// ---------------------------------------------------------------------------
__global__ __launch_bounds__(256) void wt_pack_kernel(
    const float* __restrict__ f1w, const float* __restrict__ f2w,
    const float* __restrict__ taw, const float* __restrict__ tbw,
    bf16* __restrict__ Wt) {
  int idx = blockIdx.x * 256 + threadIdx.x;       // 512*256 elements
  if (idx >= NCOL * KROW) return;
  int n = idx >> 8;          // 0..511 (output column)
  int k = idx & 255;         // 0..255 (input row)
  const float* W = (n < 128) ? f1w : (n < 256) ? f2w : (n < 384) ? taw : tbw;
  int nn = n & 127;
  Wt[n * KROW + k] = (bf16)W[k * H_DIM + nn];
}

// ---------------------------------------------------------------------------
// Kernel 2: feat = SiLU(LN(LN(x) @ proj_w + proj_b)) -> bf16 [B*T, 128]
// One row (b,t) per 128-thread block; memory-bound (~134MB total traffic,
// ~6us at 23.3 TB/s).
// ---------------------------------------------------------------------------
__global__ __launch_bounds__(128) void feat_kernel(
    const float* __restrict__ x,
    const float* __restrict__ g_in, const float* __restrict__ b_in,
    const float* __restrict__ pw,   const float* __restrict__ pb,
    const float* __restrict__ gp,   const float* __restrict__ bp,
    bf16* __restrict__ feat) {
  __shared__ float xsh[D_DIM];
  __shared__ float xnsh[D_DIM];
  __shared__ float red1[H_DIM];
  __shared__ float red2[H_DIM];

  const long row = blockIdx.x;           // b*T + t
  const int t = threadIdx.x;             // 0..127
  const float* xr = x + row * D_DIM;

  if (t < D_DIM) xsh[t] = xr[t];
  __syncthreads();

  // LN over D=67
  float s1 = 0.f, s2 = 0.f;
  if (t < D_DIM) { float v = xsh[t]; s1 = v; s2 = v * v; }
  red1[t] = s1; red2[t] = s2;
  __syncthreads();
  for (int off = 64; off > 0; off >>= 1) {
    if (t < off) { red1[t] += red1[t + off]; red2[t] += red2[t + off]; }
    __syncthreads();
  }
  float mu = red1[0] * (1.f / D_DIM);
  float var = red2[0] * (1.f / D_DIM) - mu * mu;
  float rs = rsqrtf(var + 1e-5f);
  if (t < D_DIM) xnsh[t] = (xsh[t] - mu) * rs * g_in[t] + b_in[t];
  __syncthreads();

  // projection: column t, K=67 (proj_w is [67,128] row-major -> coalesced)
  float acc = pb[t];
#pragma unroll 1
  for (int d = 0; d < D_DIM; ++d) acc += xnsh[d] * pw[d * H_DIM + t];

  // LN over H=128
  red1[t] = acc; red2[t] = acc * acc;
  __syncthreads();
  for (int off = 64; off > 0; off >>= 1) {
    if (t < off) { red1[t] += red1[t + off]; red2[t] += red2[t + off]; }
    __syncthreads();
  }
  float mu2 = red1[0] * (1.f / H_DIM);
  float var2 = red2[0] * (1.f / H_DIM) - mu2 * mu2;
  float rs2 = rsqrtf(var2 + 1e-5f);
  float y = (acc - mu2) * rs2 * gp[t] + bp[t];
  float si = y * fast_sigmoid(y);          // SiLU, branch-free
  feat[row * H_DIM + t] = (bf16)si;
}

// ---------------------------------------------------------------------------
// Kernel 3: persistent CfC scan. 4 workgroups x 512 threads (16 waves).
// Each WG owns 16 batch rows; per step computes [16,256]x[256,512] with
// v_wmma_f32_16x16x32_bf16 (wave: 2 N-tiles x 8 k-steps = 16 WMMA). The
// feat_{t+1} tile is double-buffered in LDS and filled by ASYNCcnt-tracked
// global_load_async_to_lds_b64 issued before the WMMA phase, so the copy
// overlaps the whole step. Weights stay resident in VGPRs for all 4096 steps.
// Gate phase is branch-free (v_exp_f32/v_rcp_f32) to keep the serial
// dependency chain short.
// ---------------------------------------------------------------------------
__global__ __launch_bounds__(512) void cfc_scan_kernel(
    const bf16* __restrict__ feat, const bf16* __restrict__ Wt,
    const float* __restrict__ dt,
    const float* __restrict__ f1b, const float* __restrict__ f2b,
    const float* __restrict__ tab, const float* __restrict__ tbb,
    const float* __restrict__ headw, const float* __restrict__ headb,
    float* __restrict__ out) {
  __shared__ __align__(32) bf16 cat[2][16][KROW]; // double-buffered [feat|h] 16 KB
  __shared__ float outl[16][NCOL];                // GEMM result             32 KB
  __shared__ float bias[NCOL];
  __shared__ float hw[H_DIM];

  const int tid  = threadIdx.x;
  const int wave = tid >> 5;        // 0..15
  const int lane = tid & 31;
  const int hf   = lane >> 4;       // half of the wave
  const int colw = lane & 15;       // column-in-tile / A row
  const int b0   = blockIdx.x * 16; // batch rows [b0, b0+16)
  const int r    = wave;            // gate-phase row (one wave per row)
  const int j0   = lane * 4;        // gate-phase columns [j0, j0+4)

  if (tid < H_DIM) {
    bias[tid]       = f1b[tid];
    bias[128 + tid] = f2b[tid];
    bias[256 + tid] = tab[tid];
    bias[384 + tid] = tbb[tid];
    hw[tid]         = headw[tid];
  }
  // h_0 = 0, feat_0 staged into buffer 0
  for (int jj = 0; jj < 4; ++jj) {
    int j = j0 + jj;
    cat[0][r][128 + j] = (bf16)0.f;
    cat[0][r][j] = feat[((long)(b0 + r) * T_DIM + 0) * H_DIM + j];
  }

  // Resident B-fragments: wave owns tiles {wave, wave+16} of the 512 cols.
  v16bf Bf[2][8];
#pragma unroll
  for (int ti = 0; ti < 2; ++ti) {
    int n = wave + 16 * ti;
    int col = 16 * n + colw;
#pragma unroll
    for (int kk = 0; kk < 8; ++kk)
      Bf[ti][kk] = *(const v16bf*)&Wt[col * KROW + 32 * kk + 16 * hf];
  }
  const float hb = headb[0];
  __syncthreads();

  for (int t = 0; t < T_DIM; ++t) {
    const int cur = t & 1, nxt = cur ^ 1;

    // ---- kick async copy of feat_{t+1} into the next buffer (8 B/lane) ----
    if (t + 1 < T_DIM) {
      unsigned lds_a = (unsigned)(uintptr_t)&cat[nxt][r][j0];
      async_load_b64(lds_a, &feat[((long)(b0 + r) * T_DIM + (t + 1)) * H_DIM + j0]);
    }
    if (t + 2 < T_DIM && lane == 0)        // HBM->L2 prefetch for t+2 tile
      __builtin_prefetch(&feat[((long)(b0 + r) * T_DIM + (t + 2)) * H_DIM], 0, 1);

    // ---- WMMA phase: D = cat @ W (f32 accum); preload all 8 A-frags ----
    v16bf afr[8];
#pragma unroll
    for (int kk = 0; kk < 8; ++kk) {
      // A frag: row=colw, elems[0..15]=K[32kk+16hf..)
      *((v8bf*)&afr[kk])     = *(const v8bf*)&cat[cur][colw][32 * kk + 16 * hf];
      *((v8bf*)&afr[kk] + 1) = *(const v8bf*)&cat[cur][colw][32 * kk + 16 * hf + 8];
    }
    v8f acc0 = {}, acc1 = {};
#pragma unroll
    for (int kk = 0; kk < 8; ++kk) {
      acc0 = __builtin_amdgcn_wmma_f32_16x16x32_bf16(false, afr[kk], false, Bf[0][kk],
                                                     (short)0, acc0, false, false);
      acc1 = __builtin_amdgcn_wmma_f32_16x16x32_bf16(false, afr[kk], false, Bf[1][kk],
                                                     (short)0, acc1, false, false);
    }
    // C layout: VGPR g -> row g+8*hf, col = colw
#pragma unroll
    for (int g = 0; g < 8; ++g) {
      int ro = 8 * hf + g;
      outl[ro][16 * wave + colw]        = acc0[g];
      outl[ro][16 * (wave + 16) + colw] = acc1[g];
    }
    __syncthreads();

    // ---- gate phase: wave r handles batch row b0+r, 4 h-columns per lane ----
    float dtt = dt[(long)(b0 + r) * T_DIM + t] * 10.f;
    float pl = 0.f;
#pragma unroll
    for (int jj = 0; jj < 4; ++jj) {
      int j = j0 + jj;
      float v1 = outl[r][j]       + bias[j];
      float v2 = outl[r][128 + j] + bias[128 + j];
      float va = outl[r][256 + j] + bias[256 + j];
      float vb = outl[r][384 + j] + bias[384 + j];
      float gt = fast_sigmoid(va * dtt + vb);
      float hn = fast_tanh(v1) * (1.f - gt) + gt * fast_tanh(v2);
      cat[nxt][r][128 + j] = (bf16)hn;     // h_{t+1} into next buffer
      pl += hn * hw[j];                    // fused head dot
    }

    // head reduction across the wave (all 32 lanes share row r)
#pragma unroll
    for (int off = 16; off > 0; off >>= 1) pl += __shfl_xor(pl, off, 32);
    if (lane == 0) out[(long)(b0 + r) * T_DIM + t] = pl + hb;

    wait_async0();                         // our slice of feat_{t+1} landed
    __syncthreads();                       // everyone's slice visible
  }
}

// ---------------------------------------------------------------------------
extern "C" void kernel_launch(void* const* d_in, const int* in_sizes, int n_in,
                              void* d_out, int out_size, void* d_ws, size_t ws_size,
                              hipStream_t stream) {
  const float* x      = (const float*)d_in[0];
  const float* dt     = (const float*)d_in[1];
  const float* ln_in_g = (const float*)d_in[2];
  const float* ln_in_b = (const float*)d_in[3];
  const float* proj_w = (const float*)d_in[4];
  const float* proj_b = (const float*)d_in[5];
  const float* ln_p_g = (const float*)d_in[6];
  const float* ln_p_b = (const float*)d_in[7];
  const float* ff1_w  = (const float*)d_in[8];
  const float* ff1_b  = (const float*)d_in[9];
  const float* ff2_w  = (const float*)d_in[10];
  const float* ff2_b  = (const float*)d_in[11];
  const float* ta_w   = (const float*)d_in[12];
  const float* ta_b   = (const float*)d_in[13];
  const float* tb_w   = (const float*)d_in[14];
  const float* tb_b   = (const float*)d_in[15];
  const float* head_w = (const float*)d_in[16];
  const float* head_b = (const float*)d_in[17];
  float* out = (float*)d_out;

  // workspace layout: [Wt bf16 512*256 | feat bf16 B*T*128]
  bf16* Wt   = (bf16*)d_ws;
  bf16* feat = (bf16*)((char*)d_ws + (size_t)NCOL * KROW * sizeof(bf16));

  wt_pack_kernel<<<(NCOL * KROW + 255) / 256, 256, 0, stream>>>(
      ff1_w, ff2_w, ta_w, tb_w, Wt);
  feat_kernel<<<B_DIM * T_DIM, 128, 0, stream>>>(
      x, ln_in_g, ln_in_b, proj_w, proj_b, ln_p_g, ln_p_b, feat);
  cfc_scan_kernel<<<B_DIM / 16, 512, 0, stream>>>(
      feat, Wt, dt, ff1_b, ff2_b, ta_b, tb_b, head_w, head_b, out);
}
